// DOSPNet_56616258896180
// MI455X (gfx1250) — compile-verified, matrix operand order checked by reading, not run on hardware
//
#include <hip/hip_runtime.h>
#include <hip/hip_bf16.h>

typedef __attribute__((ext_vector_type(16))) _Float16 v16h;
typedef __attribute__((ext_vector_type(8)))  _Float16 v8h;
typedef __attribute__((ext_vector_type(8)))  float    v8f;
typedef __attribute__((ext_vector_type(4)))  int      v4i;

#define WMMA_F16(A,B,C) __builtin_amdgcn_wmma_f32_16x16x32_f16(false,(A),false,(B),(short)0,(C),false,false)

#if __has_builtin(__builtin_amdgcn_global_load_async_to_lds_b128)
#define USE_ASYNC_LDS 1
#else
#define USE_ASYNC_LDS 0
#endif

__device__ __forceinline__ void wait_async0() {
#if __has_builtin(__builtin_amdgcn_s_wait_asynccnt)
  __builtin_amdgcn_s_wait_asynccnt(0);
#else
  asm volatile("s_wait_asynccnt 0" ::: "memory");
#endif
}

// ---------------------------------------------------------------------------
// f32 -> f16 conversion
// ---------------------------------------------------------------------------
__global__ void cvt_f32_f16_kernel(const float* __restrict__ src,
                                   _Float16* __restrict__ dst, size_t n) {
  size_t i = (size_t)blockIdx.x * blockDim.x + threadIdx.x;
  if (i < n) dst[i] = (_Float16)src[i];
}

// ---------------------------------------------------------------------------
// Implicit-GEMM conv + fused BN (+residual) (+ReLU), f16 in / f16 out, f32 acc
//   GEMM: M = Cout, N = B*OH*OW, K = Cin*KH*KW
//   Block: 256 threads (8 waves), macro-tile 64(M) x 128(N), K-step 32
//   Double-buffered LDS; A-tile staged with GLOBAL_LOAD_ASYNC_TO_LDS_B128
//   when in-range & 16B-aligned (block-uniform predicate, EXEC stays ~0).
// ---------------------------------------------------------------------------
struct ConvParams {
  const _Float16* act;            // [B][Cin][H][W]
  const _Float16* wgt;            // [Cout][Cin*KH*KW]
  const float *g, *b, *m, *v;     // BN params per out-channel
  const _Float16* resid;          // [B][Cout][OH][OW] or null
  _Float16* out;                  // [B][Cout][OH][OW]
  int NB, Cin, H, W, Cout, OH, OW, KH, KW, stride, pad, relu;
};

__global__ __launch_bounds__(256) void conv_wmma_kernel(ConvParams p) {
  // rows padded to 40 halves (80B, multiple of 16B) so b128 LDS ops stay aligned
  __shared__ _Float16 As[2][64][40];    // weights tile, M-major, K contiguous
  __shared__ _Float16 Bs[2][128][40];   // im2col tile,  N-major, K contiguous

  const int K    = p.Cin * p.KH * p.KW;
  const int Ntot = p.NB * p.OH * p.OW;
  const int mBase = blockIdx.y * 64;
  const int nBase = blockIdx.x * 128;
  const int tid   = threadIdx.x;
  const int lane  = tid & 31;
  const int wave  = tid >> 5;
  const int mWave = (wave & 1) * 32;
  const int nWave = (wave >> 1) * 32;
  // Uniform fast-path predicate: full M-tile in range, K row stride 16B-aligned.
  const bool alignedA = ((K & 7) == 0) && (mBase + 64 <= p.Cout);

  auto stageA = [&](int k0, int bi) {
    if (alignedA && (k0 + 32 <= K)) {
      // one 16B chunk per thread: 256 * 16B == 64x32 f16 tile
      int row = tid >> 2, kk = (tid & 3) << 3;
      const _Float16* gp = &p.wgt[(size_t)(mBase + row) * K + (k0 + kk)];
#if USE_ASYNC_LDS
      __builtin_amdgcn_global_load_async_to_lds_b128(
          (__attribute__((address_space(1))) v4i*)gp,
          (__attribute__((address_space(3))) v4i*)&As[bi][row][kk], 0, 0);
#else
      *(v8h*)&As[bi][row][kk] = *(const v8h*)gp;
#endif
    } else {
#pragma unroll
      for (int e = 0; e < 8; ++e) {
        int elem = e * 256 + tid;
        int row = elem >> 5, kk = elem & 31;
        int mg = mBase + row, kg = k0 + kk;
        _Float16 val = (_Float16)0.f;
        if (mg < p.Cout && kg < K) val = p.wgt[(size_t)mg * K + kg];
        As[bi][row][kk] = val;
      }
    }
  };

  auto stageB = [&](int k0, int bi) {
#pragma unroll
    for (int e = 0; e < 16; ++e) {
      int elem = e * 256 + tid;
      int nn = elem >> 5, kk = elem & 31;
      int col = nBase + nn, kg = k0 + kk;
      _Float16 val = (_Float16)0.f;
      if (col < Ntot && kg < K) {
        int ohw = p.OH * p.OW;
        int bb = col / ohw; int rem = col - bb * ohw;
        int oh = rem / p.OW; int ow = rem - oh * p.OW;
        int khw = p.KH * p.KW;
        int cc = kg / khw; int r2 = kg - cc * khw;
        int kh = r2 / p.KW; int kw = r2 - kh * p.KW;
        int ih = oh * p.stride - p.pad + kh;
        int iw = ow * p.stride - p.pad + kw;
        if (ih >= 0 && ih < p.H && iw >= 0 && iw < p.W)
          val = p.act[(((size_t)bb * p.Cin + cc) * p.H + ih) * p.W + iw];
      }
      Bs[bi][nn][kk] = val;
    }
  };

  v8f acc00 = {}, acc01 = {}, acc10 = {}, acc11 = {};

  // prologue: stage first tile into buffer 0
  stageA(0, 0);
  stageB(0, 0);
#if USE_ASYNC_LDS
  wait_async0();
#endif
  __syncthreads();

  int bi = 0;
  for (int k0 = 0; k0 < K; k0 += 32, bi ^= 1) {
    // issue next tile's staging into the other buffer (overlaps with WMMA)
    if (k0 + 32 < K) {
      stageA(k0 + 32, bi ^ 1);
      stageB(k0 + 32, bi ^ 1);
      __builtin_prefetch(&p.wgt[(size_t)(mBase + (tid >> 2)) * K + k0 + 64], 0, 1);
    }

    // --- fragment loads (CDNA5 16-bit A/B layout: lanes<16 hold K 0..7 &
    //     16..23, lanes>=16 hold K 8..15 & 24..31; two ds_load_b128 each) ---
    const int kh8 = (lane >> 4) << 3;
    const int ra  = mWave + (lane & 15);
    const int rb  = nWave + (lane & 15);
    v8h a0lo = *(const v8h*)&As[bi][ra][kh8],      a0hi = *(const v8h*)&As[bi][ra][kh8 + 16];
    v8h a1lo = *(const v8h*)&As[bi][ra + 16][kh8], a1hi = *(const v8h*)&As[bi][ra + 16][kh8 + 16];
    v8h b0lo = *(const v8h*)&Bs[bi][rb][kh8],      b0hi = *(const v8h*)&Bs[bi][rb][kh8 + 16];
    v8h b1lo = *(const v8h*)&Bs[bi][rb + 16][kh8], b1hi = *(const v8h*)&Bs[bi][rb + 16][kh8 + 16];
    v16h a0 = __builtin_shufflevector(a0lo, a0hi, 0,1,2,3,4,5,6,7,8,9,10,11,12,13,14,15);
    v16h a1 = __builtin_shufflevector(a1lo, a1hi, 0,1,2,3,4,5,6,7,8,9,10,11,12,13,14,15);
    v16h b0 = __builtin_shufflevector(b0lo, b0hi, 0,1,2,3,4,5,6,7,8,9,10,11,12,13,14,15);
    v16h b1 = __builtin_shufflevector(b1lo, b1hi, 0,1,2,3,4,5,6,7,8,9,10,11,12,13,14,15);

    acc00 = WMMA_F16(a0, b0, acc00);
    acc01 = WMMA_F16(a0, b1, acc01);
    acc10 = WMMA_F16(a1, b0, acc10);
    acc11 = WMMA_F16(a1, b1, acc11);

#if USE_ASYNC_LDS
    wait_async0();     // next buffer's async copy landed in LDS
#endif
    __syncthreads();   // everyone done reading buf bi / writing buf bi^1
  }

  // --- epilogue: BN -> (+residual) -> (ReLU) -> f16 store ---
  auto storeTile = [&](const v8f& acc, int mT, int nT) {
    int col = nBase + nWave + nT + (lane & 15);
    if (col >= Ntot) return;
    int ohw = p.OH * p.OW;
    int bb = col / ohw; int rem = col - bb * ohw;
    int oh = rem / p.OW; int ow = rem - oh * p.OW;
    int mrow = mBase + mWave + mT + ((lane >> 4) << 3);
#pragma unroll
    for (int j = 0; j < 8; ++j) {
      int mm = mrow + j;
      if (mm >= p.Cout) continue;
      float sc  = p.g[mm] * rsqrtf(p.v[mm] + 1e-5f);
      float val = (acc[j] - p.m[mm]) * sc + p.b[mm];
      size_t oi = (((size_t)bb * p.Cout + mm) * p.OH + oh) * p.OW + ow;
      if (p.resid) val += (float)p.resid[oi];
      if (p.relu)  val = fmaxf(val, 0.f);
      p.out[oi] = (_Float16)val;
    }
  };
  storeTile(acc00, 0, 0);  storeTile(acc01, 0, 16);
  storeTile(acc10, 16, 0); storeTile(acc11, 16, 16);
}

// ---------------------------------------------------------------------------
// 3x3 stride-2 pad-1 max pool, f16
// ---------------------------------------------------------------------------
__global__ void maxpool_kernel(const _Float16* __restrict__ in,
                               _Float16* __restrict__ out,
                               int NB, int C, int H, int W, int OH, int OW) {
  size_t i = (size_t)blockIdx.x * blockDim.x + threadIdx.x;
  size_t total = (size_t)NB * C * OH * OW;
  if (i >= total) return;
  int ow = (int)(i % OW); size_t t = i / OW;
  int oh = (int)(t % OH); t /= OH;
  int c  = (int)(t % C);  int b = (int)(t / C);
  float best = -3.0e38f;
  for (int kh = 0; kh < 3; ++kh) {
    int ih = oh * 2 - 1 + kh; if (ih < 0 || ih >= H) continue;
    for (int kw = 0; kw < 3; ++kw) {
      int iw = ow * 2 - 1 + kw; if (iw < 0 || iw >= W) continue;
      best = fmaxf(best, (float)in[(((size_t)b * C + c) * H + ih) * W + iw]);
    }
  }
  out[i] = (_Float16)best;
}

// ---------------------------------------------------------------------------
// Cyclic Jacobi eigensolver for symmetric 16x16 (single lane; matrices in LDS)
// ---------------------------------------------------------------------------
__device__ void jacobi16(float* A, float* V) {
  for (int i = 0; i < 16; ++i)
    for (int j = 0; j < 16; ++j) V[i * 16 + j] = (i == j) ? 1.f : 0.f;
  for (int sweep = 0; sweep < 14; ++sweep) {
    for (int pp = 0; pp < 15; ++pp) {
      for (int qq = pp + 1; qq < 16; ++qq) {
        float apq = A[pp * 16 + qq];
        if (fabsf(apq) < 1e-12f) continue;
        float app = A[pp * 16 + pp], aqq = A[qq * 16 + qq];
        float theta = 0.5f * (aqq - app) / apq;
        float tt = ((theta >= 0.f) ? 1.f : -1.f) /
                   (fabsf(theta) + sqrtf(theta * theta + 1.f));
        float c = rsqrtf(tt * tt + 1.f);
        float s = tt * c;
        for (int k = 0; k < 16; ++k) {
          float akp = A[k * 16 + pp], akq = A[k * 16 + qq];
          A[k * 16 + pp] = c * akp - s * akq;
          A[k * 16 + qq] = s * akp + c * akq;
        }
        for (int k = 0; k < 16; ++k) {
          float apk = A[pp * 16 + k], aqk = A[qq * 16 + k];
          A[pp * 16 + k] = c * apk - s * aqk;
          A[qq * 16 + k] = s * apk + c * aqk;
        }
        for (int k = 0; k < 16; ++k) {
          float vkp = V[k * 16 + pp], vkq = V[k * 16 + qq];
          V[k * 16 + pp] = c * vkp - s * vkq;
          V[k * 16 + qq] = s * vkp + c * vkq;
        }
      }
    }
  }
}

// ---------------------------------------------------------------------------
// SPD head: adaptive avg-pool(2x2) -> channel reduce(32) -> covariance ->
// W cov W^T -> eigh/clamp/rebuild -> eigh/log/rebuild -> triu -> fc1 -> fc2
// One workgroup per batch item.
// ---------------------------------------------------------------------------
__global__ __launch_bounds__(256) void head_kernel(
    const _Float16* __restrict__ act,   // [32][512][7][7]
    const float* __restrict__ reduce_w, // [32][512]
    const float* __restrict__ Wspd,     // [16][32]
    const float* __restrict__ fc1w, const float* __restrict__ fc1b,
    const float* __restrict__ fc2w, const float* __restrict__ fc2b,
    float* __restrict__ out)            // [32][3]
{
  __shared__ float pool[512 * 4];
  __shared__ float fmat[32 * 4];
  __shared__ float cov[32 * 32];
  __shared__ float tmp[16 * 32];
  __shared__ float Am[256], Vm[256], Bm[256];
  __shared__ float flat[136];
  __shared__ float h1[128];

  const int bI = blockIdx.x, tid = threadIdx.x;
  const _Float16* a = act + (size_t)bI * 512 * 49;

  // adaptive avg pool 7x7 -> 2x2 (bins rows/cols {0..3},{3..6})
  for (int t = tid; t < 2048; t += 256) {
    int c = t >> 2, bin = t & 3;
    int r0 = (bin >> 1) ? 3 : 0, c0 = (bin & 1) ? 3 : 0;
    float s = 0.f;
    for (int r = 0; r < 4; ++r)
      for (int q = 0; q < 4; ++q)
        s += (float)a[c * 49 + (r0 + r) * 7 + (c0 + q)];
    pool[c * 4 + bin] = s * (1.f / 16.f);
  }
  __syncthreads();
  // f[o][hw] = sum_c reduce_w[o][c] * pool[c][hw]
  for (int t = tid; t < 128; t += 256) {
    int o = t >> 2, hw = t & 3;
    float s = 0.f;
    for (int c = 0; c < 512; ++c) s += reduce_w[o * 512 + c] * pool[c * 4 + hw];
    fmat[o * 4 + hw] = s;
  }
  __syncthreads();
  if (tid < 32) {
    float mn = 0.25f * (fmat[tid*4] + fmat[tid*4+1] + fmat[tid*4+2] + fmat[tid*4+3]);
    for (int hw = 0; hw < 4; ++hw) fmat[tid * 4 + hw] -= mn;
  }
  __syncthreads();
  // cov = fc^T fc / (N-1) + 0.1 I   (N=4)
  for (int t = tid; t < 1024; t += 256) {
    int i = t >> 5, j = t & 31;
    float s = 0.f;
    for (int hw = 0; hw < 4; ++hw) s += fmat[i * 4 + hw] * fmat[j * 4 + hw];
    cov[t] = s * (1.f / 3.f) + ((i == j) ? 0.1f : 0.f);
  }
  __syncthreads();
  // c2 = W cov W^T  (16x16)
  for (int t = tid; t < 512; t += 256) {
    int o = t >> 5, j = t & 31;
    float s = 0.f;
    for (int i = 0; i < 32; ++i) s += Wspd[o * 32 + i] * cov[i * 32 + j];
    tmp[t] = s;
  }
  __syncthreads();
  for (int t = tid; t < 256; t += 256) {
    int o = t >> 4, q = t & 15;
    float s = 0.f;
    for (int j = 0; j < 32; ++j) s += tmp[o * 32 + j] * Wspd[q * 32 + j];
    Am[t] = s;
  }
  __syncthreads();
  if (tid == 0) jacobi16(Am, Vm);
  __syncthreads();
  // rebuild with eigenvalue clamp (re-eig)
  for (int t = tid; t < 256; t += 256) {
    int i = t >> 4, j = t & 15;
    float s = 0.f;
    for (int k = 0; k < 16; ++k) {
      float lam = fmaxf(Am[k * 16 + k], 1e-4f);
      s += Vm[i * 16 + k] * lam * Vm[j * 16 + k];
    }
    Bm[t] = s;
  }
  __syncthreads();
  for (int t = tid; t < 256; t += 256) Am[t] = Bm[t];
  __syncthreads();
  if (tid == 0) jacobi16(Am, Vm);
  __syncthreads();
  // log-Euclidean map
  for (int t = tid; t < 256; t += 256) {
    int i = t >> 4, j = t & 15;
    float s = 0.f;
    for (int k = 0; k < 16; ++k) {
      float lam = logf(fmaxf(Am[k * 16 + k], 1e-6f));
      s += Vm[i * 16 + k] * lam * Vm[j * 16 + k];
    }
    Bm[t] = s;
  }
  __syncthreads();
  // upper-triangular flatten (row-major, j>=i) -> 136
  for (int t = tid; t < 136; t += 256) {
    int i = 0, r = t;
    while (r >= 16 - i) { r -= 16 - i; ++i; }
    flat[t] = Bm[i * 16 + (i + r)];
  }
  __syncthreads();
  for (int t = tid; t < 128; t += 256) {
    float s = fc1b[t];
    for (int k = 0; k < 136; ++k) s += fc1w[t * 136 + k] * flat[k];
    h1[t] = fmaxf(s, 0.f);
  }
  __syncthreads();
  if (tid < 3) {
    float s = fc2b[tid];
    for (int k = 0; k < 128; ++k) s += fc2w[tid * 128 + k] * h1[k];
    out[bI * 3 + tid] = s;
  }
}

// ---------------------------------------------------------------------------
// Host orchestration
// ---------------------------------------------------------------------------
extern "C" void kernel_launch(void* const* d_in, const int* in_sizes, int n_in,
                              void* d_out, int out_size, void* d_ws, size_t ws_size,
                              hipStream_t stream) {
  (void)in_sizes; (void)n_in; (void)out_size; (void)ws_size;
  auto F = [&](int i) -> const float* { return (const float*)d_in[i]; };
  int idx = 0;
  const float* x = F(idx++);
  const float* conv1w = F(idx++);
  const float *bn1g = F(idx++), *bn1b = F(idx++), *bn1m = F(idx++), *bn1v = F(idx++);

  struct Blk {
    const float *c1w, *b1g, *b1b, *b1m, *b1v;
    const float *c2w, *b2g, *b2b, *b2m, *b2v;
    const float *dw, *dbg, *dbb, *dbm, *dbv;
    bool down;
  };
  Blk blk[8];
  for (int s = 0; s < 4; ++s)
    for (int b2 = 0; b2 < 2; ++b2) {
      Blk& B = blk[s * 2 + b2];
      B.c1w = F(idx++); B.b1g = F(idx++); B.b1b = F(idx++); B.b1m = F(idx++); B.b1v = F(idx++);
      B.c2w = F(idx++); B.b2g = F(idx++); B.b2b = F(idx++); B.b2m = F(idx++); B.b2v = F(idx++);
      B.down = (b2 == 0 && s > 0);   // stage0 block0: stride 1, in_c==out_c -> no downsample
      if (B.down) { B.dw = F(idx++); B.dbg = F(idx++); B.dbb = F(idx++); B.dbm = F(idx++); B.dbv = F(idx++); }
      else        { B.dw = nullptr;  B.dbg = B.dbb = B.dbm = B.dbv = nullptr; }
    }
  const float* reduce_w = F(idx++);
  const float* spdW     = F(idx++);
  const float* fc1w = F(idx++); const float* fc1b = F(idx++);
  const float* fc2w = F(idx++); const float* fc2b = F(idx++);

  // -- workspace bump allocator (halves); 256B-aligned blocks keep b128 legal --
  _Float16* wsh = (_Float16*)d_ws;
  size_t off = 0;
  auto alloc = [&](size_t n) -> _Float16* {
    _Float16* p = wsh + off;
    off += (n + 127) & ~(size_t)127;
    return p;
  };
  auto cvt = [&](const float* s, size_t n) -> const _Float16* {
    _Float16* d = alloc(n);
    cvt_f32_f16_kernel<<<dim3((unsigned)((n + 255) / 256)), dim3(256), 0, stream>>>(s, d, n);
    return d;
  };
  auto conv = [&](const _Float16* act, const _Float16* w,
                  const float* g, const float* bb, const float* mm, const float* vv,
                  const _Float16* resid, _Float16* out,
                  int Cin, int H, int W, int Cout, int OH, int OW,
                  int KH, int KW, int st, int pad, int relu) {
    ConvParams cp{act, w, g, bb, mm, vv, resid, out,
                  32, Cin, H, W, Cout, OH, OW, KH, KW, st, pad, relu};
    dim3 grid((unsigned)((32 * OH * OW + 127) / 128), (unsigned)((Cout + 63) / 64));
    conv_wmma_kernel<<<grid, dim3(256), 0, stream>>>(cp);
  };

  // activation buffers
  _Float16* x16 = alloc((size_t)32 * 3 * 224 * 224);
  _Float16* c1  = alloc((size_t)32 * 64 * 112 * 112);
  _Float16* buf[4];
  for (int i = 0; i < 4; ++i) buf[i] = alloc((size_t)32 * 64 * 56 * 56);

  // input -> f16
  { size_t n = (size_t)32 * 3 * 224 * 224;
    cvt_f32_f16_kernel<<<dim3((unsigned)((n + 255) / 256)), dim3(256), 0, stream>>>(x, x16, n); }

  // stem: conv7x7/2 + BN + ReLU, then maxpool3x3/2
  const _Float16* w0 = cvt(conv1w, (size_t)64 * 3 * 49);
  conv(x16, w0, bn1g, bn1b, bn1m, bn1v, nullptr, c1,
       3, 224, 224, 64, 112, 112, 7, 7, 2, 3, 1);
  { size_t total = (size_t)32 * 64 * 56 * 56;
    maxpool_kernel<<<dim3((unsigned)((total + 255) / 256)), dim3(256), 0, stream>>>(
        c1, buf[0], 32, 64, 112, 112, 56, 56); }

  // residual stages
  const int chans[4]   = {64, 128, 256, 512};
  const int strides[4] = {1, 2, 2, 2};
  int Cin = 64, Hcur = 56, cur = 0;
  for (int s = 0; s < 4; ++s) {
    int Cout = chans[s];
    for (int b2 = 0; b2 < 2; ++b2) {
      Blk& B = blk[s * 2 + b2];
      int st = (b2 == 0) ? strides[s] : 1;
      int Hin = Hcur, Hout = (Hin - 1) / st + 1;
      const _Float16* w1 = cvt(B.c1w, (size_t)Cout * Cin * 9);
      const _Float16* w2 = cvt(B.c2w, (size_t)Cout * Cout * 9);
      int hI = (cur + 1) & 3;
      int scI, outI;
      if (B.down) { scI = (cur + 2) & 3; outI = (cur + 3) & 3; }
      else        { scI = cur;           outI = (cur + 2) & 3; }
      // h = relu(bn1(conv1(x)))
      conv(buf[cur], w1, B.b1g, B.b1b, B.b1m, B.b1v, nullptr, buf[hI],
           Cin, Hin, Hin, Cout, Hout, Hout, 3, 3, st, 1, 1);
      // shortcut
      if (B.down) {
        const _Float16* wd = cvt(B.dw, (size_t)Cout * Cin);
        conv(buf[cur], wd, B.dbg, B.dbb, B.dbm, B.dbv, nullptr, buf[scI],
             Cin, Hin, Hin, Cout, Hout, Hout, 1, 1, st, 0, 0);
      }
      // out = relu(bn2(conv2(h)) + shortcut)
      conv(buf[hI], w2, B.b2g, B.b2b, B.b2m, B.b2v, buf[scI], buf[outI],
           Cout, Hout, Hout, Cout, Hout, Hout, 3, 3, 1, 1, 1);
      cur = outI; Cin = Cout; Hcur = Hout;
    }
  }

  // SPD head -> logits
  head_kernel<<<dim3(32), dim3(256), 0, stream>>>(
      buf[cur], reduce_w, spdW, fc1w, fc1b, fc2w, fc2b, (float*)d_out);
}
